// MultiHeadSiameseAttention_50319836840101
// MI455X (gfx1250) — matrix operand order repaired and checked by reading, not compiled, not run on hardware
//
#include <hip/hip_runtime.h>
#include <hip/hip_bf16.h>

// ---------------------------------------------------------------------------
// MI455X (gfx1250) implementation of MultiHeadSiameseAttention.
// B=4, NQ=1025, NK=1024, D=1024, H=16, C=64, DFF=1024.
// Compute-bound (~55 GFLOP vs ~70MB): all matmuls run on
// v_wmma_f32_16x16x32_f16 (f16 operands, fp32 accumulate); all
// normalizations / residuals stay fp32.
// ---------------------------------------------------------------------------

typedef _Float16 h16v __attribute__((ext_vector_type(16)));
typedef _Float16 h4v  __attribute__((ext_vector_type(4)));
typedef float    f8v  __attribute__((ext_vector_type(8)));
typedef float    f4v  __attribute__((ext_vector_type(4)));

#define NQ    1025
#define NK    1024
#define DMODEL 1024
#define HEADS 16
#define CHEAD 64
#define KT_PITCH 1040   // f16 pitch for K-transposed buffer (>= NK+1, mult of 16)

__device__ inline f8v zero8() {
  f8v z;
#pragma unroll
  for (int j = 0; j < 8; ++j) z[j] = 0.f;
  return z;
}
__device__ inline f4v zerof4() {
  f4v z; z[0] = z[1] = z[2] = z[3] = 0.f; return z;
}
__device__ inline h4v zeroh4() {
  h4v z; z[0] = z[1] = z[2] = z[3] = (_Float16)0.f; return z;
}

// A-matrix fragment (16x32 f16) from row-major LDS (base already offset to the
// wave's 16-row strip).  lanes 0-15 -> K 0..15,
// lanes 16-31 -> K 16..31, row = lane&15.
__device__ inline h16v lds_load_a(const _Float16* base, int pitch, int kbase) {
  int lane = threadIdx.x & 31;
  int m = lane & 15, hf = lane >> 4;
  const _Float16* p = base + m * pitch + kbase + hf * 16;
  h16v r;
#pragma unroll
  for (int j = 0; j < 16; ++j) r[j] = p[j];
  return r;
}

// B-matrix fragment (32x16 f16) from an N-major ("BT") LDS tile:
// BT[n][k].  b[e] = B[kbase + half*16 + e][n]  -> contiguous 16 f16 per lane.
__device__ inline h16v lds_load_bt(const _Float16* base, int pitch, int kbase) {
  int lane = threadIdx.x & 31;
  int n = lane & 15, hf = lane >> 4;
  const _Float16* p = base + n * pitch + kbase + hf * 16;
  h16v r;
#pragma unroll
  for (int e = 0; e < 16; ++e) r[e] = p[e];
  return r;
}

__device__ inline f8v wmma32(h16v a, h16v b, f8v c) {
  return __builtin_amdgcn_wmma_f32_16x16x32_f16(
      /*neg_a=*/false, a, /*neg_b=*/false, b,
      /*c_mod=*/(short)0, c, /*reuse_a=*/false, /*reuse_b=*/false);
}

// ---------------------------------------------------------------------------
// Kernel 1: Y = X @ w_qk, fused per-head l2norm, write f16 head-split layout.
// MODE 0: dst = QN [bh][tok][c]        (tokens per batch = NQ)
// MODE 1: dst = KNT [bh][c][tok]       (tokens per batch = NK, pitch KT_PITCH)
// grid = (16 heads, ceil(M/64)), block = 128 (4 waves). Wave w: rows 16w..16w+15.
// ---------------------------------------------------------------------------
template <int MODE>
__global__ __launch_bounds__(128) void proj_norm_kernel(
    const float* __restrict__ X, int M, int rowsPerBatch,
    const float* __restrict__ W, _Float16* __restrict__ dst) {
  const int AP = 40, BP = 40;
  __shared__ _Float16 sA[64 * 40];  // A tile: [row][k]     (64x32)
  __shared__ _Float16 sB[64 * 40];  // BT tile: [n][k]      (64n x 32k)

  int tid = threadIdx.x, wv = tid >> 5, lane = tid & 31;
  int n0 = blockIdx.x * 64;  // head * 64
  int m0 = blockIdx.y * 64;

  f8v acc[4];
#pragma unroll
  for (int t = 0; t < 4; ++t) acc[t] = zero8();

  for (int k0 = 0; k0 < DMODEL; k0 += 32) {
#pragma unroll
    for (int it = 0; it < 4; ++it) {             // A: 64x32 = 512 quads
      int idx = tid + it * 128;
      int r = idx >> 3, c4 = (idx & 7) << 2;
      f4v v = zerof4();
      int gr = m0 + r;
      if (gr < M) v = *reinterpret_cast<const f4v*>(X + (size_t)gr * DMODEL + k0 + c4);
      h4v hv;
#pragma unroll
      for (int e = 0; e < 4; ++e) hv[e] = (_Float16)v[e];
      *reinterpret_cast<h4v*>(&sA[r * AP + c4]) = hv;
    }
#pragma unroll
    for (int it = 0; it < 4; ++it) {             // B: 32k x 64n, transpose to BT
      int idx = tid + it * 128;
      int kk = idx >> 4, n4 = (idx & 15) << 2;
      f4v v = *reinterpret_cast<const f4v*>(W + (size_t)(k0 + kk) * DMODEL + n0 + n4);
#pragma unroll
      for (int e = 0; e < 4; ++e) sB[(n4 + e) * BP + kk] = (_Float16)v[e];
    }
    __syncthreads();
    h16v a = lds_load_a(sA + (wv * 16) * AP, AP, 0);
#pragma unroll
    for (int t = 0; t < 4; ++t) {
      h16v bf = lds_load_bt(sB + (t * 16) * BP, BP, 0);
      acc[t] = wmma32(a, bf, acc[t]);
    }
    __syncthreads();
  }

  // fused l2norm across the 64 head channels of each output row
  int hf = lane >> 4, ln = lane & 15;
  int h = blockIdx.x;
#pragma unroll
  for (int r = 0; r < 8; ++r) {
    float ss = 0.f;
#pragma unroll
    for (int t = 0; t < 4; ++t) { float v = acc[t][r]; ss += v * v; }
    ss += __shfl_xor(ss, 1, 32);
    ss += __shfl_xor(ss, 2, 32);
    ss += __shfl_xor(ss, 4, 32);
    ss += __shfl_xor(ss, 8, 32);
    float scale = 1.0f / fmaxf(sqrtf(ss), 1e-12f);
    int rowG = m0 + 16 * wv + hf * 8 + r;
    if (rowG < M) {
      int bb = rowG / rowsPerBatch;
      int tok = rowG - bb * rowsPerBatch;
      int bh = bb * HEADS + h;
#pragma unroll
      for (int t = 0; t < 4; ++t) {
        int col = 16 * t + ln;
        _Float16 vv = (_Float16)(acc[t][r] * scale);
        if (MODE == 0)
          dst[((size_t)bh * NQ + tok) * CHEAD + col] = vv;
        else
          dst[((size_t)bh * CHEAD + col) * KT_PITCH + tok] = vv;
      }
    }
  }
}

// ---------------------------------------------------------------------------
// Kernel 2: normalized dummy row -> KNT[:, :, NK].  grid=16 heads, block=32.
// ---------------------------------------------------------------------------
__global__ __launch_bounds__(32) void dummy_norm_kernel(
    const float* __restrict__ dummy, _Float16* __restrict__ knt) {
  int h = blockIdx.x, lane = threadIdx.x;
  float v0 = dummy[h * 64 + lane];
  float v1 = dummy[h * 64 + 32 + lane];
  float ss = v0 * v0 + v1 * v1;
  ss += __shfl_xor(ss, 1, 32);
  ss += __shfl_xor(ss, 2, 32);
  ss += __shfl_xor(ss, 4, 32);
  ss += __shfl_xor(ss, 8, 32);
  ss += __shfl_xor(ss, 16, 32);
  float scale = 1.0f / fmaxf(sqrtf(ss), 1e-12f);
  _Float16 a = (_Float16)(v0 * scale);
  _Float16 b = (_Float16)(v1 * scale);
  for (int bb = 0; bb < 4; ++bb) {
    size_t base = (size_t)(bb * HEADS + h) * CHEAD;
    knt[(base + lane) * KT_PITCH + NK] = a;
    knt[(base + 32 + lane) * KT_PITCH + NK] = b;
  }
}

// ---------------------------------------------------------------------------
// Kernel 3: attention  O = relu(Q̂ K̂ᵀ) V  per batch-head.
// grid = (ceil(NQ/64)=17, BH=64), block = 128.
// ---------------------------------------------------------------------------
__global__ __launch_bounds__(128) void attn_kernel(
    const _Float16* __restrict__ QN, const _Float16* __restrict__ KNT,
    const float* __restrict__ V, float* __restrict__ OUT) {
  const int P = 72;
  __shared__ _Float16 sQ[64 * 72];  // A: [qrow][c]
  __shared__ _Float16 sK[64 * 72];  // BT for S: [j][c]
  __shared__ _Float16 sS[64 * 72];  // A for O:  [qrow][j]
  __shared__ _Float16 sV[64 * 72];  // BT for O: [c][j]

  int tid = threadIdx.x, wv = tid >> 5, lane = tid & 31;
  int hf = lane >> 4, ln = lane & 15;
  int bh = blockIdx.y, b = bh >> 4, h = bh & 15;
  int i0 = blockIdx.x * 64;

#pragma unroll
  for (int it = 0; it < 8; ++it) {               // Q tile 64x64 f16
    int idx = tid + it * 128;
    int row = idx >> 4, c4 = (idx & 15) << 2;
    h4v v = zeroh4();
    if (i0 + row < NQ)
      v = *reinterpret_cast<const h4v*>(QN + ((size_t)bh * NQ + i0 + row) * CHEAD + c4);
    *reinterpret_cast<h4v*>(&sQ[row * P + c4]) = v;
  }

  f8v accO[4];
#pragma unroll
  for (int t = 0; t < 4; ++t) accO[t] = zero8();

  for (int j0 = 0; j0 <= NK; j0 += 64) {         // 17 chunks (last: 1 valid row)
#pragma unroll
    for (int it = 0; it < 8; ++it) {             // K chunk -> sK[j][c] (transpose)
      int idx = tid + it * 128;
      int c = idx >> 4, j4 = (idx & 15) << 2;
      const _Float16* src = KNT + ((size_t)bh * CHEAD + c) * KT_PITCH + j0 + j4;
#pragma unroll
      for (int e = 0; e < 4; ++e) {
        _Float16 kv = (_Float16)0.f;
        if (j0 + j4 + e <= NK) kv = src[e];
        sK[(j4 + e) * P + c] = kv;
      }
    }
#pragma unroll
    for (int it = 0; it < 8; ++it) {             // V chunk -> sV[c][j] (transpose)
      int idx = tid + it * 128;
      int j = idx >> 4, c4 = (idx & 15) << 2;
      f4v v = zerof4();
      int jj = j0 + j;
      if (jj < NK)
        v = *reinterpret_cast<const f4v*>(V + ((size_t)b * NK + jj) * DMODEL + h * CHEAD + c4);
#pragma unroll
      for (int e = 0; e < 4; ++e) sV[(c4 + e) * P + j] = (_Float16)v[e];
    }
    __syncthreads();

    f8v accS[4];
#pragma unroll
    for (int t = 0; t < 4; ++t) accS[t] = zero8();
#pragma unroll
    for (int ks = 0; ks < 64; ks += 32) {
      h16v a = lds_load_a(sQ + (wv * 16) * P, P, ks);
#pragma unroll
      for (int t = 0; t < 4; ++t)
        accS[t] = wmma32(a, lds_load_bt(sK + (t * 16) * P, P, ks), accS[t]);
    }
#pragma unroll
    for (int t = 0; t < 4; ++t)
#pragma unroll
      for (int r = 0; r < 8; ++r)
        sS[(16 * wv + hf * 8 + r) * P + 16 * t + ln] = (_Float16)fmaxf(accS[t][r], 0.f);
    __syncthreads();

#pragma unroll
    for (int ks = 0; ks < 64; ks += 32) {
      h16v a = lds_load_a(sS + (wv * 16) * P, P, ks);
#pragma unroll
      for (int t = 0; t < 4; ++t)
        accO[t] = wmma32(a, lds_load_bt(sV + (t * 16) * P, P, ks), accO[t]);
    }
    __syncthreads();
  }

#pragma unroll
  for (int r = 0; r < 8; ++r) {
    int i = i0 + 16 * wv + hf * 8 + r;
    if (i < NQ) {
#pragma unroll
      for (int t = 0; t < 4; ++t)
        OUT[((size_t)bh * NQ + i) * CHEAD + 16 * t + ln] = accO[t][r];
    }
  }
}

// ---------------------------------------------------------------------------
// Kernel 4: per-head mixer + FFN + LayerNorm + vh + merge-heads store.
// grid = (17, 64), block = 128.
// ---------------------------------------------------------------------------
__global__ __launch_bounds__(128) void epilogue_kernel(
    const float* __restrict__ OUT, const float* __restrict__ Q,
    const float* __restrict__ V,
    const float* __restrict__ w1p, const float* __restrict__ b1p,
    const float* __restrict__ w2p, const float* __restrict__ b2p,
    const float* __restrict__ w3p, const float* __restrict__ b3p,
    const float* __restrict__ fw1p, const float* __restrict__ fb1p,
    const float* __restrict__ fw2p, const float* __restrict__ fb2p,
    const float* __restrict__ lng, const float* __restrict__ lnb,
    float* __restrict__ out) {
  const int P = 136;
  __shared__ _Float16 sA[64 * 136];  // Xa (64x128); later Ta (cols 0..63) + W2t (cols 68..131)
  __shared__ _Float16 sB[64 * 136];  // weights BT
  __shared__ _Float16 sC[64 * 136];  // Xb = [o1,o2 | rh]; later Hh

  int tid = threadIdx.x, wv = tid >> 5, lane = tid & 31;
  int hf = lane >> 4, ln = lane & 15;
  int bh = blockIdx.y, b = bh >> 4, h = bh & 15;
  int i0 = blockIdx.x * 64;

  // ---- Phase 0: build Xa = [out*rh | rh-out]; pre-fill Xb high half with rh
#pragma unroll
  for (int it = 0; it < 8; ++it) {
    int idx = tid + it * 128;
    int row = idx >> 4, c4 = (idx & 15) << 2;
    int i = i0 + row;
    f4v o = zerof4(), r_ = zerof4();
    if (i < NQ) {
      o  = *reinterpret_cast<const f4v*>(OUT + ((size_t)bh * NQ + i) * CHEAD + c4);
      r_ = *reinterpret_cast<const f4v*>(Q + ((size_t)b * NQ + i) * DMODEL + h * CHEAD + c4);
    }
#pragma unroll
    for (int e = 0; e < 4; ++e) {
      sA[row * P + c4 + e]      = (_Float16)(o[e] * r_[e]);
      sA[row * P + 64 + c4 + e] = (_Float16)(r_[e] - o[e]);
      sC[row * P + 64 + c4 + e] = (_Float16)r_[e];
    }
  }
  // block-diag weight BT: [n][k]: n<32,k<64 -> w1[k][n] ; n>=32,k>=64 -> w2[k-64][n-32]
#pragma unroll
  for (int it = 0; it < 16; ++it) {
    int idx = tid + it * 128;
    int n = idx >> 5, k4 = (idx & 31) << 2;
    h4v hv;
#pragma unroll
    for (int e = 0; e < 4; ++e) {
      int k = k4 + e;
      float w = 0.f;
      if (n < 32 && k < 64)             w = w1p[k * 32 + n];
      else if (n >= 32 && k >= 64)      w = w2p[(k - 64) * 32 + (n - 32)];
      hv[e] = (_Float16)w;
    }
    *reinterpret_cast<h4v*>(&sB[n * P + k4]) = hv;
  }
  __syncthreads();

  // ---- GEMM1: [o1|o2] = relu(Xa @ Wd + [b1,b2])
  f8v acc1[4];
#pragma unroll
  for (int t = 0; t < 4; ++t) acc1[t] = zero8();
#pragma unroll
  for (int ks = 0; ks < 128; ks += 32) {
    h16v a = lds_load_a(sA + (wv * 16) * P, P, ks);
#pragma unroll
    for (int t = 0; t < 4; ++t)
      acc1[t] = wmma32(a, lds_load_bt(sB + (t * 16) * P, P, ks), acc1[t]);
  }
#pragma unroll
  for (int t = 0; t < 4; ++t) {
    int col = 16 * t + ln;
    float bias = (col < 32) ? b1p[col] : b2p[col - 32];
#pragma unroll
    for (int r = 0; r < 8; ++r)
      sC[(16 * wv + hf * 8 + r) * P + col] = (_Float16)fmaxf(acc1[t][r] + bias, 0.f);
  }
  __syncthreads();
  // w3ᵀ -> sB : [n][k] = w3[k][n], k<128, n<64
#pragma unroll
  for (int it = 0; it < 16; ++it) {
    int idx = tid + it * 128;
    int k = idx >> 4, n4 = (idx & 15) << 2;
    f4v v = *reinterpret_cast<const f4v*>(w3p + (size_t)k * 64 + n4);
#pragma unroll
    for (int e = 0; e < 4; ++e) sB[(n4 + e) * P + k] = (_Float16)v[e];
  }
  __syncthreads();

  // ---- GEMM2: o = Xb @ w3 + b3 ; t = o + rh
  f8v acc2[4];
#pragma unroll
  for (int t = 0; t < 4; ++t) acc2[t] = zero8();
#pragma unroll
  for (int ks = 0; ks < 128; ks += 32) {
    h16v a = lds_load_a(sC + (wv * 16) * P, P, ks);
#pragma unroll
    for (int t = 0; t < 4; ++t)
      acc2[t] = wmma32(a, lds_load_bt(sB + (t * 16) * P, P, ks), acc2[t]);
  }
  float treg[4][8];
#pragma unroll
  for (int t = 0; t < 4; ++t) {
    int col = 16 * t + ln;
    float b3v = b3p[col];
#pragma unroll
    for (int r = 0; r < 8; ++r) {
      int i = i0 + 16 * wv + hf * 8 + r;
      float rv = (i < NQ) ? Q[((size_t)b * NQ + i) * DMODEL + h * CHEAD + col] : 0.f;
      treg[t][r] = acc2[t][r] + b3v + rv;
    }
  }
  __syncthreads();

  // ---- FFN: t + relu(t@fw1+fb1)@fw2 (+fb2 folded in at the end)
#pragma unroll
  for (int t = 0; t < 4; ++t)
#pragma unroll
    for (int r = 0; r < 8; ++r)
      sA[(16 * wv + hf * 8 + r) * P + 16 * t + ln] = (_Float16)treg[t][r];

  f8v accF[4];
#pragma unroll
  for (int t = 0; t < 4; ++t) accF[t] = zero8();
  __syncthreads();

  for (int hc = 0; hc < 16; ++hc) {
#pragma unroll
    for (int it = 0; it < 8; ++it) {   // fw1 column chunk -> sB[n][k]
      int idx = tid + it * 128;
      int k = idx >> 4, n4 = (idx & 15) << 2;
      f4v v = *reinterpret_cast<const f4v*>(fw1p + (size_t)k * 1024 + hc * 64 + n4);
#pragma unroll
      for (int e = 0; e < 4; ++e) sB[(n4 + e) * P + k] = (_Float16)v[e];
    }
#pragma unroll
    for (int it = 0; it < 8; ++it) {   // fw2 row chunk -> sA cols 68..131
      int idx = tid + it * 128;
      int k = idx >> 4, n4 = (idx & 15) << 2;
      f4v v = *reinterpret_cast<const f4v*>(fw2p + ((size_t)(hc * 64 + k)) * 64 + n4);
#pragma unroll
      for (int e = 0; e < 4; ++e) sA[(n4 + e) * P + 68 + k] = (_Float16)v[e];
    }
    __syncthreads();

    f8v acc3[4];
#pragma unroll
    for (int t = 0; t < 4; ++t) acc3[t] = zero8();
#pragma unroll
    for (int ks = 0; ks < 64; ks += 32) {
      h16v a = lds_load_a(sA + (wv * 16) * P, P, ks);
#pragma unroll
      for (int t = 0; t < 4; ++t)
        acc3[t] = wmma32(a, lds_load_bt(sB + (t * 16) * P, P, ks), acc3[t]);
    }
#pragma unroll
    for (int t = 0; t < 4; ++t) {
      int col = 16 * t + ln;
      float bias = fb1p[hc * 64 + col];
#pragma unroll
      for (int r = 0; r < 8; ++r)
        sC[(16 * wv + hf * 8 + r) * P + col] = (_Float16)fmaxf(acc3[t][r] + bias, 0.f);
    }
    __syncthreads();

#pragma unroll
    for (int ks = 0; ks < 64; ks += 32) {
      h16v a = lds_load_a(sC + (wv * 16) * P, P, ks);
#pragma unroll
      for (int t = 0; t < 4; ++t)
        accF[t] = wmma32(a, lds_load_bt(sA + (t * 16) * P + 68, P, ks), accF[t]);
    }
    __syncthreads();
  }

  // ---- LayerNorm over 64 channels + vh + store (merge heads)
#pragma unroll
  for (int r = 0; r < 8; ++r) {
    float v0 = treg[0][r] + accF[0][r] + fb2p[ln];
    float v1 = treg[1][r] + accF[1][r] + fb2p[16 + ln];
    float v2 = treg[2][r] + accF[2][r] + fb2p[32 + ln];
    float v3 = treg[3][r] + accF[3][r] + fb2p[48 + ln];
    float s = v0 + v1 + v2 + v3;
    s += __shfl_xor(s, 1, 32);
    s += __shfl_xor(s, 2, 32);
    s += __shfl_xor(s, 4, 32);
    s += __shfl_xor(s, 8, 32);
    float mu = s * (1.f / 64.f);
    float d0 = v0 - mu, d1 = v1 - mu, d2 = v2 - mu, d3 = v3 - mu;
    float q2 = d0 * d0 + d1 * d1 + d2 * d2 + d3 * d3;
    q2 += __shfl_xor(q2, 1, 32);
    q2 += __shfl_xor(q2, 2, 32);
    q2 += __shfl_xor(q2, 4, 32);
    q2 += __shfl_xor(q2, 8, 32);
    float inv = rsqrtf(q2 * (1.f / 64.f) + 1e-5f);
    int i = i0 + 16 * wv + hf * 8 + r;
    if (i < NQ) {
      float dd[4] = {d0, d1, d2, d3};
#pragma unroll
      for (int t = 0; t < 4; ++t) {
        int col = 16 * t + ln;
        float y = dd[t] * inv * lng[col] + lnb[col];
        float vh = (i < NK) ? V[((size_t)b * NK + i) * DMODEL + h * CHEAD + col] : 0.f;
        out[((size_t)b * NQ + i) * DMODEL + h * CHEAD + col] = y + vh;
      }
    }
  }
}

// ---------------------------------------------------------------------------
extern "C" void kernel_launch(void* const* d_in, const int* in_sizes, int n_in,
                              void* d_out, int out_size, void* d_ws, size_t ws_size,
                              hipStream_t stream) {
  (void)in_sizes; (void)n_in; (void)out_size; (void)ws_size;
  const float* q     = (const float*)d_in[0];
  const float* k     = (const float*)d_in[1];
  const float* v     = (const float*)d_in[2];
  const float* w_qk  = (const float*)d_in[3];
  const float* dummy = (const float*)d_in[4];
  const float* w1    = (const float*)d_in[5];
  const float* b1    = (const float*)d_in[6];
  const float* w2    = (const float*)d_in[7];
  const float* b2    = (const float*)d_in[8];
  const float* w3    = (const float*)d_in[9];
  const float* b3    = (const float*)d_in[10];
  const float* fw1   = (const float*)d_in[11];
  const float* fb1   = (const float*)d_in[12];
  const float* fw2   = (const float*)d_in[13];
  const float* fb2   = (const float*)d_in[14];
  const float* ln_g  = (const float*)d_in[15];
  const float* ln_b  = (const float*)d_in[16];
  float* outp = (float*)d_out;

  // workspace carve (~34 MB)
  char* ws = (char*)d_ws;
  _Float16* QN = (_Float16*)ws;                                   // 64*1025*64 f16
  size_t off = (((size_t)64 * NQ * CHEAD * 2) + 255) & ~(size_t)255;
  _Float16* KNT = (_Float16*)(ws + off);                          // 64*64*1040 f16
  off += (((size_t)64 * CHEAD * KT_PITCH * 2) + 255) & ~(size_t)255;
  float* OUTb = (float*)(ws + off);                               // 64*1025*64 f32

  // 1) q @ w_qk -> normalized Q heads (M = B*NQ = 4100 -> 65 row tiles)
  proj_norm_kernel<0><<<dim3(16, 65), 128, 0, stream>>>(q, 4 * NQ, NQ, w_qk, QN);
  // 2) k @ w_qk -> normalized K heads, transposed (M = B*NK = 4096)
  proj_norm_kernel<1><<<dim3(16, 64), 128, 0, stream>>>(k, 4 * NK, NK, w_qk, KNT);
  // 3) normalized dummy key row
  dummy_norm_kernel<<<16, 32, 0, stream>>>(dummy, KNT);
  // 4) attention
  attn_kernel<<<dim3(17, 64), 128, 0, stream>>>(QN, KNT, v, OUTb);
  // 5) mixer + FFN + LN + vh + merge heads
  epilogue_kernel<<<dim3(17, 64), 128, 0, stream>>>(
      OUTb, q, v, w1, b1, w2, b2, w3, b3, fw1, fb1, fw2, fb2, ln_g, ln_b, outp);
}
